// HydraChannelMixer_14594298872042
// MI455X (gfx1250) — matrix operand order repaired and checked by reading, not compiled, not run on hardware
//
#include <hip/hip_runtime.h>

// Problem dims (fixed by the reference)
#define D_MODEL 512
#define C_CH    64
#define P_PATCH 128
#define B_SIZE  8
#define NGROUP  (B_SIZE * P_PATCH)   // 1024 independent (b,p) groups
#define NTHREADS 256
#define NWAVES   8                   // wave32

typedef __attribute__((ext_vector_type(16))) __bf16 v16bf;
typedef __attribute__((ext_vector_type(8)))  __bf16 v8bf;
typedef __attribute__((ext_vector_type(8)))  float  v8f;

static __device__ __forceinline__ unsigned short f2bf(float f) {
    unsigned int u = __builtin_bit_cast(unsigned int, f);
    u += 0x7FFFu + ((u >> 16) & 1u);          // round-to-nearest-even
    return (unsigned short)(u >> 16);
}
static __device__ __forceinline__ float bf2f(unsigned short h) {
    unsigned int u = ((unsigned int)h) << 16;
    return __builtin_bit_cast(float, u);
}
static __device__ __forceinline__ unsigned int pack2(float a, float b) {
    return (unsigned int)f2bf(a) | ((unsigned int)f2bf(b) << 16);
}

// ---------------------------------------------------------------------------
// Pre-pass: convert Wq|Wk|Wv (f32, row-major [e][d]) to bf16 into workspace.
// 3 * 512 * 512 * 2B = 1.5 MB of d_ws; stays resident in the 192MB L2.
// ---------------------------------------------------------------------------
__global__ void hydra_convert_weights(const float* __restrict__ Wq,
                                      const float* __restrict__ Wk,
                                      const float* __restrict__ Wv,
                                      unsigned short* __restrict__ out) {
    const int n = D_MODEL * D_MODEL;
    int i = blockIdx.x * blockDim.x + threadIdx.x;
    if (i < n)            out[i] = f2bf(Wq[i]);
    else if (i < 2 * n)   out[i] = f2bf(Wk[i - n]);
    else if (i < 3 * n)   out[i] = f2bf(Wv[i - 2 * n]);
}

// ---------------------------------------------------------------------------
// Fused Q/K/V GEMM: 64x512 = (64x512 bf16 A in LDS) x (512x512 bf16 W^T).
// Wave w owns output cols [64w, 64w+64).  Loop nest: col-tile -> k -> row-slab
// with 12 resident accumulator tiles (Q,K,V x 4 row-slabs), so per k-step:
//   8 ds_load_b128 (4 A frags, shared by Q/K/V) + 6 global_load_b128 (3 B
//   frags, each reused across 4 row-slabs)  ->  12 v_wmma_f32_16x16x32_bf16.
// Fragment layouts used here (contiguous k = 16h + s; A and B match):
//  A (16x32): lane -> M = lane&15, Kb = 16*(lane>>4); halfs = K[Kb..Kb+7],
//             K[Kb+8..Kb+15] (two contiguous 16B LDS runs).
//  B (32x16): lane -> N = lane&15, Kr = 16*(lane>>4); halfs = W[e0+N][Kr..Kr+15]
//             (one contiguous 32B run: B = W^T needs no transpose).
//  C/D (16x16 f32): vgpr r -> M = r + 8*(lane>>4), N = lane&15.
// ---------------------------------------------------------------------------
static __device__ __forceinline__ void gemm_qkv(const unsigned short* __restrict__ hn,
                                                const unsigned short* __restrict__ wq,
                                                const unsigned short* __restrict__ wk,
                                                const unsigned short* __restrict__ wv,
                                                unsigned short* __restrict__ qs,
                                                unsigned short* __restrict__ ks,
                                                unsigned short* __restrict__ vs,
                                                int lane, int wave) {
    const int e0w = wave * 64;
    const int mA  = lane & 15;
    const int kbA = (lane >> 4) * 16;
    const int eB  = lane & 15;
    const int krB = (lane >> 4) * 16;
    for (int ct = 0; ct < 4; ++ct) {
        const int e0   = e0w + ct * 16;
        const int boff = (e0 + eB) * D_MODEL + krB;
        v8f aq[4] = {};
        v8f ak[4] = {};
        v8f av[4] = {};
#pragma unroll 2
        for (int kk = 0; kk < 16; ++kk) {
            const int k0 = kk * 32;
            v16bf bq = *(const v16bf*)(wq + boff + k0);   // 2x global_load_b128
            v16bf bk = *(const v16bf*)(wk + boff + k0);
            v16bf bv = *(const v16bf*)(wv + boff + k0);
#pragma unroll
            for (int rs = 0; rs < 4; ++rs) {
                const unsigned short* arow = hn + (rs * 16 + mA) * D_MODEL + kbA + k0;
                v8bf alo = *(const v8bf*)(arow);          // ds_load_b128
                v8bf ahi = *(const v8bf*)(arow + 8);      // ds_load_b128
                v16bf af = __builtin_shufflevector(alo, ahi,
                    0, 1, 2, 3, 4, 5, 6, 7, 8, 9, 10, 11, 12, 13, 14, 15);
                aq[rs] = __builtin_amdgcn_wmma_f32_16x16x32_bf16(
                    false, af, false, bq, (short)0, aq[rs], false, false);
                ak[rs] = __builtin_amdgcn_wmma_f32_16x16x32_bf16(
                    false, af, false, bk, (short)0, ak[rs], false, false);
                av[rs] = __builtin_amdgcn_wmma_f32_16x16x32_bf16(
                    false, af, false, bv, (short)0, av[rs], false, false);
            }
        }
        const int eC  = e0 + (lane & 15);
        const int mB8 = (lane >> 4) * 8;
#pragma unroll
        for (int rs = 0; rs < 4; ++rs) {
            const int mC = rs * 16 + mB8;
#pragma unroll
            for (int r = 0; r < 8; ++r) {
                qs[(mC + r) * D_MODEL + eC] = f2bf(aq[rs][r]);
                ks[(mC + r) * D_MODEL + eC] = f2bf(ak[rs][r]);
                vs[(mC + r) * D_MODEL + eC] = f2bf(av[rs][r]);
            }
        }
    }
}

// ---------------------------------------------------------------------------
// One workgroup per (b,p) group.  LDS: hn/qs/ks/vs 64x512 bf16 (256 KB) +
// gamma/beta/gf f32 + per-row inv-norms  ->  ~262 KB of the 320 KB WGP LDS.
// ---------------------------------------------------------------------------
__launch_bounds__(NTHREADS, 1)
__global__ void hydra_channel_mixer(const float* __restrict__ x,
                                    const float* __restrict__ gamma,
                                    const float* __restrict__ beta,
                                    const float* __restrict__ alphap,
                                    const unsigned short* __restrict__ wb,
                                    float* __restrict__ out) {
    extern __shared__ char smem_raw[];
    unsigned short* hn = (unsigned short*)smem_raw;        // 64x512 bf16
    unsigned short* qs = hn + C_CH * D_MODEL;
    unsigned short* ks = qs + C_CH * D_MODEL;
    unsigned short* vs = ks + C_CH * D_MODEL;
    float* rq  = (float*)(vs + C_CH * D_MODEL);            // [64] 1/||Q_row||
    float* rk  = rq + C_CH;                                // [64] 1/||K_row||
    float* gf  = rk + C_CH;                                // [512]
    float* gls = gf + D_MODEL;                             // gamma [512]
    float* bls = gls + D_MODEL;                            // beta  [512]

    const int tid  = threadIdx.x;
    const int lane = tid & 31;
    const int wave = tid >> 5;
    const int n    = blockIdx.x;            // group id = b*P + p
    const int b    = n >> 7;                // / P_PATCH
    const int p    = n & (P_PATCH - 1);
    const float alpha = *alphap;

    // stage gamma/beta
    gls[tid]       = gamma[tid];
    gls[tid + 256] = gamma[tid + 256];
    bls[tid]       = beta[tid];
    bls[tid + 256] = beta[tid + 256];
    __syncthreads();

    // ---- Phase A: gather + LayerNorm -> hn (bf16) ----
    for (int c = wave; c < C_CH; c += NWAVES) {
        const float* xr = x + (((size_t)(b * C_CH + c)) * P_PATCH + p) * D_MODEL;
        float4 v[4];
        float s = 0.f, ss = 0.f;
#pragma unroll
        for (int i = 0; i < 4; ++i) {
            v[i] = ((const float4*)xr)[lane + 32 * i];
            s  += v[i].x + v[i].y + v[i].z + v[i].w;
            ss += v[i].x * v[i].x + v[i].y * v[i].y + v[i].z * v[i].z + v[i].w * v[i].w;
        }
        for (int off = 16; off; off >>= 1) {
            s  += __shfl_xor(s,  off, 32);
            ss += __shfl_xor(ss, off, 32);
        }
        const float mu  = s * (1.f / D_MODEL);
        const float var = ss * (1.f / D_MODEL) - mu * mu;
        const float rs  = rsqrtf(var + 1e-5f);
#pragma unroll
        for (int i = 0; i < 4; ++i) {
            const int d0 = 4 * (lane + 32 * i);
            const float t0 = (v[i].x - mu) * rs * gls[d0 + 0] + bls[d0 + 0];
            const float t1 = (v[i].y - mu) * rs * gls[d0 + 1] + bls[d0 + 1];
            const float t2 = (v[i].z - mu) * rs * gls[d0 + 2] + bls[d0 + 2];
            const float t3 = (v[i].w - mu) * rs * gls[d0 + 3] + bls[d0 + 3];
            uint2 pk; pk.x = pack2(t0, t1); pk.y = pack2(t2, t3);
            *(uint2*)(hn + c * D_MODEL + d0) = pk;           // ds_store_b64
        }
    }
    __syncthreads();

    // ---- Phase B: fused Q/K/V GEMMs via v_wmma_f32_16x16x32_bf16 ----
    gemm_qkv(hn,
             wb,
             wb + D_MODEL * D_MODEL,
             wb + 2 * D_MODEL * D_MODEL,
             qs, ks, vs, lane, wave);
    __syncthreads();

    // ---- Phase C: per-row inverse L2 norms of Q and K ----
    for (int c = wave; c < C_CH; c += NWAVES) {
        float sq = 0.f, sk = 0.f;
#pragma unroll
        for (int i = 0; i < 4; ++i) {
            uint2 q4 = ((const uint2*)(qs + c * D_MODEL))[lane + 32 * i];
            uint2 k4 = ((const uint2*)(ks + c * D_MODEL))[lane + 32 * i];
            float a, bb_;
            a = bf2f((unsigned short)(q4.x & 0xffff)); bb_ = bf2f((unsigned short)(q4.x >> 16));
            sq += a * a + bb_ * bb_;
            a = bf2f((unsigned short)(q4.y & 0xffff)); bb_ = bf2f((unsigned short)(q4.y >> 16));
            sq += a * a + bb_ * bb_;
            a = bf2f((unsigned short)(k4.x & 0xffff)); bb_ = bf2f((unsigned short)(k4.x >> 16));
            sk += a * a + bb_ * bb_;
            a = bf2f((unsigned short)(k4.y & 0xffff)); bb_ = bf2f((unsigned short)(k4.y >> 16));
            sk += a * a + bb_ * bb_;
        }
        for (int off = 16; off; off >>= 1) {
            sq += __shfl_xor(sq, off, 32);
            sk += __shfl_xor(sk, off, 32);
        }
        if (lane == 0) {
            rq[c] = 1.f / fmaxf(sqrtf(sq), 1e-12f);
            rk[c] = 1.f / fmaxf(sqrtf(sk), 1e-12f);
        }
    }
    __syncthreads();

    // ---- Phase D: gf[d] = sum_c K_norm[c,d] * V[c,d]  (2 d's per thread) ----
    {
        const int d0 = 2 * tid;
        float a0 = 0.f, a1 = 0.f;
#pragma unroll 8
        for (int c = 0; c < C_CH; ++c) {
            const unsigned int k2 = *(const unsigned int*)(ks + c * D_MODEL + d0);
            const unsigned int v2 = *(const unsigned int*)(vs + c * D_MODEL + d0);
            const float rkc = rk[c];
            a0 += bf2f((unsigned short)(k2 & 0xffff)) * rkc * bf2f((unsigned short)(v2 & 0xffff));
            a1 += bf2f((unsigned short)(k2 >> 16))    * rkc * bf2f((unsigned short)(v2 >> 16));
        }
        gf[d0]     = a0;
        gf[d0 + 1] = a1;
    }
    __syncthreads();

    // ---- Phase E: out = h + alpha * Q_norm * gf  (scatter back to x layout)
    for (int c = wave; c < C_CH; c += NWAVES) {
        const size_t off = (((size_t)(b * C_CH + c)) * P_PATCH + p) * D_MODEL;
        const float aq = alpha * rq[c];
#pragma unroll
        for (int i = 0; i < 4; ++i) {
            const int idx = lane + 32 * i;
            const int d0  = idx * 4;
            float4 hv = ((const float4*)(x + off))[idx];
            float4 o;
            o.x = hv.x + bf2f(qs[c * D_MODEL + d0 + 0]) * aq * gf[d0 + 0];
            o.y = hv.y + bf2f(qs[c * D_MODEL + d0 + 1]) * aq * gf[d0 + 1];
            o.z = hv.z + bf2f(qs[c * D_MODEL + d0 + 2]) * aq * gf[d0 + 2];
            o.w = hv.w + bf2f(qs[c * D_MODEL + d0 + 3]) * aq * gf[d0 + 3];
            ((float4*)(out + off))[idx] = o;
        }
    }
}

extern "C" void kernel_launch(void* const* d_in, const int* in_sizes, int n_in,
                              void* d_out, int out_size, void* d_ws, size_t ws_size,
                              hipStream_t stream) {
    (void)in_sizes; (void)n_in; (void)out_size; (void)ws_size;
    const float* x     = (const float*)d_in[0];
    const float* Wq    = (const float*)d_in[1];
    const float* Wk    = (const float*)d_in[2];
    const float* Wv    = (const float*)d_in[3];
    const float* gamma = (const float*)d_in[4];
    const float* beta  = (const float*)d_in[5];
    const float* alpha = (const float*)d_in[6];
    unsigned short* wbf = (unsigned short*)d_ws;       // 1.5 MB bf16 weights

    const int nconv = 3 * D_MODEL * D_MODEL;
    hydra_convert_weights<<<(nconv + 255) / 256, 256, 0, stream>>>(Wq, Wk, Wv, wbf);

    // LDS: 4 * 64*512*2  +  (64+64+512+512+512)*4  = 268800 B  (< 320 KB/WGP)
    const size_t lds_bytes =
        4u * C_CH * D_MODEL * sizeof(unsigned short) +
        (2u * C_CH + 3u * D_MODEL) * sizeof(float);
    hydra_channel_mixer<<<NGROUP, NTHREADS, lds_bytes, stream>>>(
        x, gamma, beta, alpha, wbf, (float*)d_out);
}
